// ASSETAttention_45277545234672
// MI455X (gfx1250) — matrix operand order repaired and checked by reading, not compiled, not run on hardware
//
#include <hip/hip_runtime.h>
#include <hip/hip_bf16.h>
#include <math.h>

typedef __attribute__((ext_vector_type(16))) _Float16 v16h;
typedef __attribute__((ext_vector_type(8)))  _Float16 v8h;
typedef __attribute__((ext_vector_type(8)))  float    v8f;
typedef __attribute__((ext_vector_type(4)))  float    v4f;
typedef int v4i_vs __attribute__((vector_size(16)));   // matches builtin's <4 x i32>

#define B_   2
#define S_   4096
#define E_   1024
#define H_   16
#define D_   64
#define NB_  64
#define BS_  64
#define NR_  3
#define SCALE_ 0.125f   // D^-0.5 = 1/8

// CDNA5 async global->LDS copy (ASYNCcnt-tracked), if the toolchain exposes it
#if defined(__has_builtin)
#if __has_builtin(__builtin_amdgcn_global_load_async_to_lds_b128) && \
    __has_builtin(__builtin_amdgcn_s_wait_asynccnt)
#define USE_ASYNC_LDS 1
#endif
#endif

static __device__ __forceinline__ v16h cat8(v8h lo, v8h hi) {
  v16h r;
#pragma unroll
  for (int j = 0; j < 8; ++j) { r[j] = lo[j]; r[8 + j] = hi[j]; }
  return r;
}

// ---------------------------------------------------------------------------
// Kernel 0: f32 -> f16 bulk convert (hidden and the three weight matrices).
// ---------------------------------------------------------------------------
__global__ __launch_bounds__(256)
void cvt_f32_to_f16_kernel(const float* __restrict__ src, _Float16* __restrict__ dst,
                           int n8)
{
  const int i = blockIdx.x * blockDim.x + threadIdx.x;
  if (i >= n8) return;
  const v4f* p = reinterpret_cast<const v4f*>(src + (size_t)i * 8);
  v4f x0 = p[0], x1 = p[1];
  v8h o;
#pragma unroll
  for (int j = 0; j < 4; ++j) { o[j] = (_Float16)x0[j]; o[4 + j] = (_Float16)x1[j]; }
  *reinterpret_cast<v8h*>(dst + (size_t)i * 8) = o;
}

// ---------------------------------------------------------------------------
// Kernel 1: QKV projection.  out = X @ W.T + b  (TN GEMM, M=8192, N=K=1024),
// X and W pre-converted to f16.  W 64x32 chunk staged in LDS per block; with
// async-to-LDS support the stage is double-buffered so the DMA of chunk i+1
// overlaps the 4 WMMAs of chunk i.  Q,K written f16 (B,H,S,D); V written f16
// transposed (B,H,D,S).  grid = (64, 16, 3): 8 waves, wave -> 16(M) x 64(N).
// ---------------------------------------------------------------------------
__global__ __launch_bounds__(256)
void sga_proj_kernel(const _Float16* __restrict__ Xh,
                     const _Float16* __restrict__ Whq,
                     const _Float16* __restrict__ Whk,
                     const _Float16* __restrict__ Whv,
                     const float* __restrict__ bq, const float* __restrict__ bk,
                     const float* __restrict__ bv,
                     _Float16* __restrict__ Qh, _Float16* __restrict__ Kh,
                     _Float16* __restrict__ VTh)
{
  __shared__ _Float16 wtile[2][64 * 32];   // double-buffered 64(N) x 32(K) chunk

  const int lane = threadIdx.x & 31;
  const int wave = threadIdx.x >> 5;
  const int lm   = lane & 15;     // A row / B,C column within tile
  const int g    = lane >> 4;

  const int mtile = blockIdx.x * 8 + wave;   // 0..511 (16 rows each)
  const int nbase = blockIdx.y * 64;         // N strip base
  const int pz    = blockIdx.z;              // 0=Q 1=K 2=V

  const _Float16* W    = (pz == 0) ? Whq : (pz == 1) ? Whk : Whv;
  const float*    bias = (pz == 0) ? bq  : (pz == 1) ? bk  : bv;

  const int       mrow = mtile * 16 + lm;    // global M for A rows
  const _Float16* xrow = Xh + (size_t)mrow * E_;

  // cooperative W-stage indices: thread -> (row 0..63, 8-col quarter 0..3)
  const int wr = threadIdx.x >> 2;
  const int wq = threadIdx.x & 3;
  const _Float16* wsrc = W + (size_t)(nbase + wr) * E_ + wq * 8;

  v8f acc[4] = {v8f{}, v8f{}, v8f{}, v8f{}};

  // one 32-K chunk of the GEMM against a staged W buffer
  auto compute = [&](const _Float16* wt, int kc) {
    v8h xa = *reinterpret_cast<const v8h*>(xrow + kc + g * 8);
    v8h xb = *reinterpret_cast<const v8h*>(xrow + kc + 16 + g * 8);
    v16h a = cat8(xa, xb);
#pragma unroll
    for (int t = 0; t < 4; ++t) {
      const _Float16* wrow = wt + (t * 16 + lm) * 32 + g * 8;
      v8h w0 = *reinterpret_cast<const v8h*>(wrow);
      v8h w1 = *reinterpret_cast<const v8h*>(wrow + 16);
      acc[t] = __builtin_amdgcn_wmma_f32_16x16x32_f16(false, a, false, cat8(w0, w1),
                                                      (short)0, acc[t], false, false);
    }
  };

#ifdef USE_ASYNC_LDS
  // fire-and-forget 16B/thread async copy of chunk ci into buffer nb
  auto stage = [&](int ci, int nb) {
    v4i_vs* gsrc = reinterpret_cast<v4i_vs*>(const_cast<_Float16*>(wsrc + ci * 32));
    v4i_vs* ldst = reinterpret_cast<v4i_vs*>(&wtile[nb][wr * 32 + wq * 8]);
    __builtin_amdgcn_global_load_async_to_lds_b128(
        (__attribute__((address_space(1))) v4i_vs*)gsrc,
        (__attribute__((address_space(3))) v4i_vs*)ldst,
        0, 0);
  };
  stage(0, 0);                               // prologue: chunk 0
  for (int ci = 0; ci < E_ / 32; ++ci) {
    __builtin_amdgcn_s_wait_asynccnt(0);     // my async stores to LDS landed
    __syncthreads();                         // everyone's landed; prev compute done
    if (ci + 1 < E_ / 32) stage(ci + 1, (ci + 1) & 1);  // overlap DMA with WMMAs
    compute(wtile[ci & 1], ci * 32);
  }
#else
  for (int ci = 0; ci < E_ / 32; ++ci) {
    *reinterpret_cast<v8h*>(&wtile[0][wr * 32 + wq * 8]) =
        *reinterpret_cast<const v8h*>(wsrc + ci * 32);
    __syncthreads();
    compute(wtile[0], ci * 32);
    __syncthreads();
  }
#endif

  // bias + store.  C layout: lane holds N = lm (fixed), M = 8*g + rr.
#pragma unroll
  for (int t = 0; t < 4; ++t) {
    const int   ncol = nbase + t * 16 + lm;
    const float bv_  = bias[ncol];
    const int   h    = ncol >> 6;        // head
    const int   d    = ncol & 63;        // dim in head
    if (pz < 2) {
      _Float16* dst = (pz == 0) ? Qh : Kh;
#pragma unroll
      for (int rr = 0; rr < 8; ++rr) {
        const int m = mtile * 16 + 8 * g + rr;
        const int bb = m >> 12, s = m & (S_ - 1);
        dst[(((size_t)bb * H_ + h) * S_ + s) * D_ + d] = (_Float16)(acc[t][rr] + bv_);
      }
    } else {
      // V transposed: (B,H,D,S) -> lane's 8 M-values are contiguous in S
      const int m0 = mtile * 16 + 8 * g;
      const int bb = m0 >> 12, s0 = m0 & (S_ - 1);
      v8h pack;
#pragma unroll
      for (int rr = 0; rr < 8; ++rr) pack[rr] = (_Float16)(acc[t][rr] + bv_);
      *reinterpret_cast<v8h*>(VTh + (((size_t)bb * H_ + h) * D_ + d) * S_ + s0) = pack;
    }
  }
}

// ---------------------------------------------------------------------------
// Kernel 2: block-sparse attention.  One 256-thread block per (b,h,qblock).
// Scores (64 x nkeys) f32 in LDS, 4-thread/row softmax stats, P fragments
// re-materialized from LDS with exp + normalization fused in.
// ---------------------------------------------------------------------------
__global__ __launch_bounds__(256)
void sga_attn_kernel(const _Float16* __restrict__ Qh, const _Float16* __restrict__ Kh,
                     const _Float16* __restrict__ VTh, const int* __restrict__ rand_attn,
                     float* __restrict__ out)
{
  extern __shared__ float smem[];
  float* scores = smem;                     // 64 * 384
  float* red    = smem + 64 * 384;          // 64 * 4
  float* rowmax = red + 64 * 4;             // 64
  float* rowinv = rowmax + 64;              // 64

  const int n  = blockIdx.x;
  const int h  = blockIdx.y;
  const int b  = blockIdx.z;
  const int bh = b * H_ + h;

  const int lane = threadIdx.x & 31;
  const int wave = threadIdx.x >> 5;
  const int lm   = lane & 15;
  const int g    = lane >> 4;

  int blist[6];
  int nk;
  const int* ra = rand_attn + ((size_t)bh * NB_ + n) * NR_;
  if (n == 0)            { blist[0] = 0;      blist[1] = 1;      blist[2] = ra[0]; blist[3] = ra[1]; blist[4] = ra[2]; nk = 5; }
  else if (n == NB_ - 1) { blist[0] = NB_-2;  blist[1] = NB_-1;  blist[2] = ra[0]; blist[3] = ra[1]; blist[4] = ra[2]; nk = 5; }
  else                   { blist[0] = n - 1;  blist[1] = n;      blist[2] = n + 1; blist[3] = ra[0]; blist[4] = ra[1]; blist[5] = ra[2]; nk = 6; }
  const int nkeys = nk * BS_;   // 320 or 384
  const int nk4   = nk * 4;     // 16-wide key tiles

  // ---------------- QK^T ----------------
  const int qt = wave & 3;      // this wave's query 16-tile
  v16h a0, a1;
  {
    const _Float16* qrow = Qh + ((size_t)bh * S_ + n * BS_ + qt * 16 + lm) * D_;
    v8h r0 = *reinterpret_cast<const v8h*>(qrow + g * 8);
    v8h r1 = *reinterpret_cast<const v8h*>(qrow + 16 + g * 8);
    v8h r2 = *reinterpret_cast<const v8h*>(qrow + 32 + g * 8);
    v8h r3 = *reinterpret_cast<const v8h*>(qrow + 48 + g * 8);
    a0 = cat8(r0, r1);          // d = 0..31
    a1 = cat8(r2, r3);          // d = 32..63
  }

  for (int kt = (wave >> 2); kt < nk4; kt += 2) {
    const int bi   = kt >> 2;
    const int srow = blist[bi] * BS_ + (kt & 3) * 16 + lm;
    const _Float16* krow = Kh + ((size_t)bh * S_ + srow) * D_;
    v8h k0 = *reinterpret_cast<const v8h*>(krow + g * 8);
    v8h k1 = *reinterpret_cast<const v8h*>(krow + 16 + g * 8);
    v8h k2 = *reinterpret_cast<const v8h*>(krow + 32 + g * 8);
    v8h k3 = *reinterpret_cast<const v8h*>(krow + 48 + g * 8);
    v16h b0 = cat8(k0, k1), b1 = cat8(k2, k3);

    v8f acc = {};
    acc = __builtin_amdgcn_wmma_f32_16x16x32_f16(false, a0, false, b0, (short)0, acc, false, false);
    acc = __builtin_amdgcn_wmma_f32_16x16x32_f16(false, a1, false, b1, (short)0, acc, false, false);
#pragma unroll
    for (int rr = 0; rr < 8; ++rr)
      scores[(qt * 16 + 8 * g + rr) * 384 + kt * 16 + lm] = acc[rr] * SCALE_;
  }
  __syncthreads();

  // ---------------- softmax stats (4 threads per row) ----------------
  {
    const int r = threadIdx.x >> 2, part = threadIdx.x & 3;
    const int seg = nkeys >> 2;                      // 80 or 96
    const float* srow = scores + r * 384 + part * seg;
    float m = -1e30f;
    for (int j = 0; j < seg; ++j) m = fmaxf(m, srow[j]);
    red[r * 4 + part] = m;
    __syncthreads();
    const float rm = fmaxf(fmaxf(red[r * 4 + 0], red[r * 4 + 1]),
                           fmaxf(red[r * 4 + 2], red[r * 4 + 3]));
    __syncthreads();
    float s = 0.f;
    for (int j = 0; j < seg; ++j) s += __expf(srow[j] - rm);
    red[r * 4 + part] = s;
    __syncthreads();
    if (part == 0) {
      rowmax[r] = rm;
      rowinv[r] = 1.0f / (red[r * 4 + 0] + red[r * 4 + 1] + red[r * 4 + 2] + red[r * 4 + 3]);
    }
    __syncthreads();
  }

  // ---------------- P @ V ----------------
  const int   dt0  = (wave >> 2) * 2;           // this wave's two d 16-tiles
  const int   arow = qt * 16 + lm;              // P fragment row (fixed per lane)
  const float rm   = rowmax[arow];
  const float ri   = rowinv[arow];
  const _Float16* v0base = VTh + ((size_t)bh * D_ + dt0 * 16 + lm) * S_;
  const _Float16* v1base = VTh + ((size_t)bh * D_ + (dt0 + 1) * 16 + lm) * S_;

  v8f c0 = {}, c1 = {};
  const int nchunk = nkeys >> 5;                // 10 or 12
  for (int c = 0; c < nchunk; ++c) {
    // P fragment: softmax normalize fused into the load (all halves same row)
    const float* sr = scores + arow * 384 + c * 32;
    v16h a;
#pragma unroll
    for (int j = 0; j < 8; ++j) {
      a[j]     = (_Float16)(__expf(sr[g * 8 + j] - rm) * ri);
      a[8 + j] = (_Float16)(__expf(sr[16 + g * 8 + j] - rm) * ri);
    }
    const int bi    = c >> 1;
    const int sbase = blist[bi] * BS_ + (c & 1) * 32 + g * 8;
    v8h u0 = *reinterpret_cast<const v8h*>(v0base + sbase);
    v8h u1 = *reinterpret_cast<const v8h*>(v0base + sbase + 16);
    v8h w0 = *reinterpret_cast<const v8h*>(v1base + sbase);
    v8h w1 = *reinterpret_cast<const v8h*>(v1base + sbase + 16);
    c0 = __builtin_amdgcn_wmma_f32_16x16x32_f16(false, a, false, cat8(u0, u1), (short)0, c0, false, false);
    c1 = __builtin_amdgcn_wmma_f32_16x16x32_f16(false, a, false, cat8(w0, w1), (short)0, c1, false, false);
  }

  // ---------------- store ctx (B,S,E) f32 ----------------
  float* orow = out + ((size_t)b * S_ + n * BS_) * E_ + h * D_;
#pragma unroll
  for (int rr = 0; rr < 8; ++rr) {
    const int q = qt * 16 + 8 * g + rr;
    orow[(size_t)q * E_ + dt0 * 16 + lm]       = c0[rr];
    orow[(size_t)q * E_ + (dt0 + 1) * 16 + lm] = c1[rr];
  }
}

// ---------------------------------------------------------------------------
extern "C" void kernel_launch(void* const* d_in, const int* in_sizes, int n_in,
                              void* d_out, int out_size, void* d_ws, size_t ws_size,
                              hipStream_t stream)
{
  const float* hidden    = (const float*)d_in[0];
  const int*   rand_attn = (const int*)  d_in[1];
  const float* Wq = (const float*)d_in[2];
  const float* bq = (const float*)d_in[3];
  const float* Wk = (const float*)d_in[4];
  const float* bk = (const float*)d_in[5];
  const float* Wv = (const float*)d_in[6];
  const float* bv = (const float*)d_in[7];
  float* out = (float*)d_out;

  const size_t qkv_elems = (size_t)B_ * H_ * S_ * D_;   // 8,388,608 halves
  const size_t w_elems   = (size_t)E_ * E_;             // 1,048,576 halves
  _Float16* Qh  = (_Float16*)d_ws;
  _Float16* Kh  = Qh  + qkv_elems;
  _Float16* VTh = Kh  + qkv_elems;
  _Float16* Xh  = VTh + qkv_elems;                      // B*S*E halves
  _Float16* Whq = Xh  + qkv_elems;
  _Float16* Whk = Whq + w_elems;
  _Float16* Whv = Whk + w_elems;
  // total workspace: 4*16MB + 3*2MB = 70MB

  // f32 -> f16 pre-convert (all L2 resident afterwards)
  cvt_f32_to_f16_kernel<<<dim3((int)(qkv_elems / 8 / 256)), 256, 0, stream>>>(hidden, Xh, (int)(qkv_elems / 8));
  cvt_f32_to_f16_kernel<<<dim3((int)(w_elems / 8 / 256)),   256, 0, stream>>>(Wq, Whq, (int)(w_elems / 8));
  cvt_f32_to_f16_kernel<<<dim3((int)(w_elems / 8 / 256)),   256, 0, stream>>>(Wk, Whk, (int)(w_elems / 8));
  cvt_f32_to_f16_kernel<<<dim3((int)(w_elems / 8 / 256)),   256, 0, stream>>>(Wv, Whv, (int)(w_elems / 8));

  // Projections: M/16 = 512 strips, 8 waves/block -> grid.x = 64
  sga_proj_kernel<<<dim3(64, E_ / 64, 3), 256, 0, stream>>>(
      Xh, Whq, Whk, Whv, bq, bk, bv, Qh, Kh, VTh);

  const size_t smem = (64 * 384 + 64 * 4 + 64 + 64) * sizeof(float);  // ~99 KB of 320 KB LDS
  sga_attn_kernel<<<dim3(NB_, H_, B_), 256, smem, stream>>>(
      Qh, Kh, VTh, rand_attn, out);
}
